// GPN_79302276153387
// MI455X (gfx1250) — matrix operand order repaired and checked by reading, not compiled, not run on hardware
//
#include <hip/hip_runtime.h>

#define N_NODES 100000
#define D_FEAT 512
#define D_HID 64
#define D_LAT 16
#define N_CLASS 10
#define N_FLOW 10
#define K_PROP 10

typedef __attribute__((ext_vector_type(16))) __bf16 v16bf;
typedef __attribute__((ext_vector_type(8)))  float  v8f;
typedef __attribute__((ext_vector_type(4)))  unsigned tdm_u4;
typedef __attribute__((ext_vector_type(8)))  int      tdm_i8;
typedef __attribute__((ext_vector_type(4)))  int      tdm_i4;

union Frag16 {
  v16bf    v;
  unsigned u[8];
  uint4    q[2];
};

__device__ __forceinline__ unsigned fbits(float f) {
  union { float f; unsigned u; } v; v.f = f;
  return v.u;
}
// RNE f32->bf16 (used only in one-time weight prepack kernels)
__device__ __forceinline__ unsigned short f2bf_rne(float f) {
  const unsigned u = fbits(f);
  return (unsigned short)((u + 0x7FFFu + ((u >> 16) & 1u)) >> 16);
}
__device__ __forceinline__ unsigned packbf_rne(float a, float b) {
  return (unsigned)f2bf_rne(a) | ((unsigned)f2bf_rne(b) << 16);
}
// 1-VALU-op truncating bf16 pair pack: lo16 = bf16(a), hi16 = bf16(b)
__device__ __forceinline__ unsigned packbf_fast(float a, float b) {
  return __builtin_amdgcn_perm(fbits(a), fbits(b), 0x03020706u);
}
__device__ __forceinline__ float softplus_f(float x) {
  return (x > 20.0f) ? x : log1pf(expf(x));
}

// Build A fragment (16x32 bf16 WMMA layout) from 16 contiguous f32 (this lane's k half).
__device__ __forceinline__ void load_afrag(Frag16& af, const float* ap) {
  const float4 f0 = *(const float4*)(ap + 0);
  const float4 f1 = *(const float4*)(ap + 4);
  const float4 f2 = *(const float4*)(ap + 8);
  const float4 f3 = *(const float4*)(ap + 12);
  af.u[0] = packbf_fast(f0.x, f0.y); af.u[1] = packbf_fast(f0.z, f0.w);
  af.u[2] = packbf_fast(f1.x, f1.y); af.u[3] = packbf_fast(f1.z, f1.w);
  af.u[4] = packbf_fast(f2.x, f2.y); af.u[5] = packbf_fast(f2.z, f2.w);
  af.u[6] = packbf_fast(f3.x, f3.y); af.u[7] = packbf_fast(f3.z, f3.w);
}

// TDM: async-DMA 4 KB (1024 dwords) from global to LDS. D# per ISA §8.3/8.4.
// 6-arg builtin variant (clang-23 / therock-10.0 headers).
__device__ __forceinline__ void tdm_load_4kb(unsigned lds_off, const void* gptr) {
  const unsigned long long ga = (unsigned long long)(uintptr_t)gptr;
  tdm_u4 g0;
  g0[0] = 1u;                                               // count=1 (valid D#)
  g0[1] = lds_off;                                          // lds_addr
  g0[2] = (unsigned)ga;                                     // global_addr[95:64]
  g0[3] = ((unsigned)(ga >> 32) & 0x1FFFFFFu) | (2u << 30); // addr hi | type=2
  tdm_i8 g1;
  g1[0] = 0x00020000;          // wg_mask=0, data_size=2 (4 bytes)
  g1[1] = (int)(1024u << 16);  // tensor_dim0[15:0]=1024 in bits[31:16]
  g1[2] = (int)(1u << 16);     // tensor_dim0 hi=0 | tensor_dim1=1
  g1[3] = (int)(1024u << 16);  // tensor_dim1 hi=0 | tile_dim0=1024
  g1[4] = 0;                   // tile_dim1=0 (unused), tile_dim2=0
  g1[5] = 1024;                // tensor_dim0_stride
  g1[6] = 0;
  g1[7] = 0;
  const tdm_i4 z4 = {0, 0, 0, 0};
  const tdm_i8 z8 = {0, 0, 0, 0, 0, 0, 0, 0};
  __builtin_amdgcn_tensor_load_to_lds(g0, g1, z4, z4, z8, 0);
}

// ---------------------------------------------------------------------------
// One-time weight prepack into WMMA B-fragment layout (bf16 pairs, RNE).
// w1pk index: ((ks*4 + nb)*32 + lane)*8 + v ; column n = nb*16+(lane&15),
// k0 = ks*32 + (lane>>4)*16 + 2v (32x16 bf16 B layout).
// ---------------------------------------------------------------------------
__global__ void prepack_w1_kernel(const float* __restrict__ W1,
                                  unsigned* __restrict__ w1pk) {
  const int t = blockIdx.x * blockDim.x + threadIdx.x;
  if (t >= 16 * 4 * 32 * 8) return;
  const int v    = t & 7;
  const int lane = (t >> 3) & 31;
  const int nb   = (t >> 8) & 3;
  const int ks   = t >> 10;
  const int n    = nb * 16 + (lane & 15);
  const int k0   = ks * 32 + ((lane >> 4) * 16) + 2 * v;
  w1pk[t] = packbf_rne(W1[k0 * D_HID + n], W1[(k0 + 1) * D_HID + n]);
}

__global__ void prepack_w2_kernel(const float* __restrict__ W2,
                                  unsigned* __restrict__ w2pk) {
  const int t = blockIdx.x * blockDim.x + threadIdx.x;
  if (t >= 2 * 32 * 8) return;
  const int v    = t & 7;
  const int lane = (t >> 3) & 31;
  const int ks   = t >> 8;
  const int n    = lane & 15;
  const int k0   = ks * 32 + ((lane >> 4) * 16) + 2 * v;
  w2pk[t] = packbf_rne(W2[k0 * D_LAT + n], W2[(k0 + 1) * D_LAT + n]);
}

// ---------------------------------------------------------------------------
// z = relu(x@W1 + b1) @ W2 + b2, fused. 8 waves/block; each wave owns two
// 16-row M tiles (block = 256 rows). W1 K-slabs DMA'd to LDS by the TDM,
// double-buffered so the DMA overlaps the 8 WMMAs of the current K-step.
// ---------------------------------------------------------------------------
__global__ void encode_kernel(const float* __restrict__ x,
                              const unsigned* __restrict__ w1pk,
                              const float* __restrict__ bias1,
                              const unsigned* __restrict__ w2pk,
                              const float* __restrict__ bias2,
                              float* __restrict__ z) {
  __shared__ unsigned slab[2][1024];   // double-buffered W1 fragment slab (8 KB)
  __shared__ float hbuf[8][16][68];    // per-wave h tile staging, padded (34 KB)

  const int tid   = threadIdx.x;
  const int wave  = tid >> 5;
  const int lane  = tid & 31;
  const int lhalf = lane >> 4;
  const int l15   = lane & 15;
  const int m0    = blockIdx.x * 256 + wave * 32;

  int rowA0 = m0 + l15;        if (rowA0 > N_NODES - 1) rowA0 = N_NODES - 1;
  int rowA1 = m0 + 16 + l15;   if (rowA1 > N_NODES - 1) rowA1 = N_NODES - 1;
  const float* xrow0 = x + (size_t)rowA0 * D_FEAT;
  const float* xrow1 = x + (size_t)rowA1 * D_FEAT;

  // Prologue: DMA K-step 0 slab, wait, release.
  if (wave == 0) {
    tdm_load_4kb((unsigned)(uintptr_t)&slab[0][0], w1pk);
    __builtin_amdgcn_s_wait_tensorcnt(0);
  }
  __syncthreads();

  const v8f zero8 = {0.f, 0.f, 0.f, 0.f, 0.f, 0.f, 0.f, 0.f};
  v8f acc[2][4] = {{zero8, zero8, zero8, zero8}, {zero8, zero8, zero8, zero8}};

  for (int ks = 0; ks < D_FEAT / 32; ++ks) {
    const int cur = ks & 1;
    const int nxt = cur ^ 1;
    // Overlap: DMA the next slab while this K-step computes.
    if (wave == 0 && ks + 1 < D_FEAT / 32) {
      tdm_load_4kb((unsigned)(uintptr_t)&slab[nxt][0], w1pk + (ks + 1) * 1024);
    }

    Frag16 af0, af1;
    load_afrag(af0, xrow0 + ks * 32 + lhalf * 16);
    load_afrag(af1, xrow1 + ks * 32 + lhalf * 16);

#pragma unroll
    for (int nb = 0; nb < 4; ++nb) {
      Frag16 bf_;
      const uint4* bp = (const uint4*)&slab[cur][nb * 256 + lane * 8];
      bf_.q[0] = bp[0];
      bf_.q[1] = bp[1];
      acc[0][nb] = __builtin_amdgcn_wmma_f32_16x16x32_bf16(
          false, af0.v, false, bf_.v, (short)0, acc[0][nb], false, false);
      acc[1][nb] = __builtin_amdgcn_wmma_f32_16x16x32_bf16(
          false, af1.v, false, bf_.v, (short)0, acc[1][nb], false, false);
    }

    __syncthreads();                              // everyone done with slab[cur]
    if (wave == 0) __builtin_amdgcn_s_wait_tensorcnt(0);  // slab[nxt] landed
    __syncthreads();                              // release next slab
  }

  // Stage 2 per M tile: bias+relu -> hbuf -> z = h @ W2 + b2.
  for (int t = 0; t < 2; ++t) {
    __syncthreads();   // previous tile's hbuf reads complete
#pragma unroll
    for (int nb = 0; nb < 4; ++nb) {
      const int n = nb * 16 + l15;
      const float bb = bias1[n];
#pragma unroll
      for (int r = 0; r < 8; ++r) {
        const float hv = acc[t][nb][r] + bb;   // C layout: M = r + 8*lhalf
        hbuf[wave][r + lhalf * 8][n] = hv > 0.f ? hv : 0.f;
      }
    }
    __syncthreads();

    v8f acc2 = zero8;
#pragma unroll
    for (int ks2 = 0; ks2 < 2; ++ks2) {
      Frag16 af;
      load_afrag(af, &hbuf[wave][l15][ks2 * 32 + lhalf * 16]);
      Frag16 bf_;
      const uint4* wp = (const uint4*)(w2pk + (ks2 * 32 + lane) * 8);
      bf_.q[0] = wp[0];
      bf_.q[1] = wp[1];
      acc2 = __builtin_amdgcn_wmma_f32_16x16x32_bf16(
          false, af.v, false, bf_.v, (short)0, acc2, false, false);
    }

    const float bb2 = bias2[l15];
#pragma unroll
    for (int r = 0; r < 8; ++r) {
      const int row = m0 + t * 16 + r + lhalf * 8;
      if (row < N_NODES) z[(size_t)row * D_LAT + l15] = acc2[r] + bb2;
    }
  }
}

// ---------------------------------------------------------------------------
// Per-(node,class) radial-flow log prob -> beta_ft (writes x0 and b)
// ---------------------------------------------------------------------------
__global__ void flow_kernel(const float* __restrict__ z,
                            const float* __restrict__ fz0,
                            const float* __restrict__ fap,
                            const float* __restrict__ fbeta,
                            const float* __restrict__ p_c,
                            float* __restrict__ x0,
                            float* __restrict__ bprop) {
  const int t = blockIdx.x * blockDim.x + threadIdx.x;
  if (t >= N_NODES * N_CLASS) return;
  const int i = t / N_CLASS;
  const int c = t - i * N_CLASS;

  float zz[D_LAT];
  {
    const float4* zp = (const float4*)(z + (size_t)i * D_LAT);
#pragma unroll
    for (int q4 = 0; q4 < 4; ++q4) {
      const float4 v = zp[q4];
      zz[q4 * 4 + 0] = v.x; zz[q4 * 4 + 1] = v.y;
      zz[q4 * 4 + 2] = v.z; zz[q4 * 4 + 3] = v.w;
    }
  }

  float sum_logdet = 0.f;
#pragma unroll
  for (int l = 0; l < N_FLOW; ++l) {
    const float* z0p = fz0 + ((l * N_CLASS + c) * D_LAT);
    const float alpha    = softplus_f(fap[l * N_CLASS + c]);
    const float beta_hat = -alpha + softplus_f(fbeta[l * N_CLASS + c]);
    float dz[D_LAT];
    float r2 = 0.f;
#pragma unroll
    for (int d = 0; d < D_LAT; ++d) { dz[d] = zz[d] - z0p[d]; r2 += dz[d] * dz[d]; }
    const float r  = sqrtf(r2);
    const float h  = 1.f / (alpha + r);
    const float bh = beta_hat * h;
#pragma unroll
    for (int d = 0; d < D_LAT; ++d) zz[d] += bh * dz[d];
    sum_logdet += (float)(D_LAT - 1) * log1pf(bh) +
                  log1pf(bh - beta_hat * r * h * h);
  }

  float q = 0.f;
#pragma unroll
  for (int d = 0; d < D_LAT; ++d) q += zz[d] * zz[d];
  const float log_base = -0.5f * q - 14.7030165f;                  // 0.5*D*log(2pi)
  float val = log_base + sum_logdet + logf(p_c[c]) + 20.2481940f;  // +0.5*D*log(4pi)
  val = fminf(fmaxf(val, -30.f), 30.f);
  const float bft = expf(val);
  x0[t]    = bft;
  bprop[t] = bft;
}

// ---------------------------------------------------------------------------
// APPNP (edges/b/agg are L2-resident: 25.6 MB + 4 MB + 4 MB << 192 MB)
// ---------------------------------------------------------------------------
__global__ void zero_kernel(float* __restrict__ p, int n) {
  const int t = blockIdx.x * blockDim.x + threadIdx.x;
  if (t < n) p[t] = 0.f;
}

__global__ void degree_kernel(const int* __restrict__ dst,
                              float* __restrict__ deg, int E) {
  const int t = blockIdx.x * blockDim.x + threadIdx.x;
  if (t < E) atomicAdd(&deg[dst[t]], 1.0f);
}

__global__ void dinv_kernel(float* __restrict__ dinv) {
  const int t = blockIdx.x * blockDim.x + threadIdx.x;
  if (t < N_NODES) dinv[t] = rsqrtf(dinv[t] + 1.0f);  // +1 self loop, always > 0
}

__global__ void scatter_kernel(const int* __restrict__ src,
                               const int* __restrict__ dst,
                               const float* __restrict__ dinv,
                               const float* __restrict__ b,
                               float* __restrict__ agg, int E) {
  const int t = blockIdx.x * blockDim.x + threadIdx.x;
  if (t >= E) return;
  if (t + 2048 < E) {  // stream-ahead prefetch on the edge list
    __builtin_prefetch(&src[t + 2048], 0, 1);
    __builtin_prefetch(&dst[t + 2048], 0, 1);
  }
  const int s = src[t], d = dst[t];
  const float nrm = dinv[s] * dinv[d];
  const float* bs = b + (size_t)s * N_CLASS;
  float* ad = agg + (size_t)d * N_CLASS;
#pragma unroll
  for (int c = 0; c < N_CLASS; ++c) atomicAdd(&ad[c], nrm * bs[c]);
}

__global__ void combine_kernel(const float* __restrict__ agg,
                               const float* __restrict__ dinv,
                               const float* __restrict__ x0,
                               float* __restrict__ b) {
  const int t = blockIdx.x * blockDim.x + threadIdx.x;
  if (t >= N_NODES * N_CLASS) return;
  const int i = t / N_CLASS;
  const float di = dinv[i];
  const float a = agg[t] + di * di * b[t];  // self-loop folded in analytically
  b[t] = 0.9f * a + 0.1f * x0[t];
}

__global__ void finalize_kernel(const float* __restrict__ b,
                                float* __restrict__ out) {
  const int i = blockIdx.x * blockDim.x + threadIdx.x;
  if (i >= N_NODES) return;
  float a[N_CLASS];
  float s = 0.f;
#pragma unroll
  for (int c = 0; c < N_CLASS; ++c) { a[c] = 1.f + b[(size_t)i * N_CLASS + c]; s += a[c]; }
  const float inv = 1.f / s;
  int best = 0;
  float bv = a[0];
#pragma unroll
  for (int c = 0; c < N_CLASS; ++c) {
    out[N_NODES + (size_t)i * N_CLASS + c] = a[c] * inv;
    if (a[c] > bv) { bv = a[c]; best = c; }
  }
  out[i] = (float)best;
}

// ---------------------------------------------------------------------------
extern "C" void kernel_launch(void* const* d_in, const int* in_sizes, int n_in,
                              void* d_out, int out_size, void* d_ws, size_t ws_size,
                              hipStream_t stream) {
  const float* x     = (const float*)d_in[0];
  const int*   edge  = (const int*)d_in[1];
  const float* p_c   = (const float*)d_in[2];
  const float* W1    = (const float*)d_in[3];
  const float* bias1 = (const float*)d_in[4];
  const float* W2    = (const float*)d_in[5];
  const float* bias2 = (const float*)d_in[6];
  const float* fz0   = (const float*)d_in[7];
  const float* fap   = (const float*)d_in[8];
  const float* fbeta = (const float*)d_in[9];

  const int E = in_sizes[1] / 2;
  const int* src = edge;
  const int* dst = edge + E;

  // Workspace (floats): z[N*16] | x0[N*10] | b[N*10] | agg[N*10] | dinv[N] | packed weights
  float*    ws   = (float*)d_ws;
  float*    z    = ws;
  float*    x0   = ws + 1600000;
  float*    b    = ws + 2600000;
  float*    agg  = ws + 3600000;
  float*    dinv = ws + 4600000;
  unsigned* w1pk = (unsigned*)(ws + 4700032);   // 16384 dwords, 256B-aligned
  unsigned* w2pk = w1pk + 16384;                // 512 dwords

  const dim3 blk(256);

  prepack_w1_kernel<<<(16384 + 255) / 256, blk, 0, stream>>>(W1, w1pk);
  prepack_w2_kernel<<<(512 + 255) / 256, blk, 0, stream>>>(W2, w2pk);

  encode_kernel<<<(N_NODES + 255) / 256, blk, 0, stream>>>(x, w1pk, bias1, w2pk, bias2, z);

  flow_kernel<<<(N_NODES * N_CLASS + 255) / 256, blk, 0, stream>>>(
      z, fz0, fap, fbeta, p_c, x0, b);

  zero_kernel<<<(N_NODES + 255) / 256, blk, 0, stream>>>(dinv, N_NODES);
  degree_kernel<<<(E + 255) / 256, blk, 0, stream>>>(dst, dinv, E);
  dinv_kernel<<<(N_NODES + 255) / 256, blk, 0, stream>>>(dinv);

  for (int it = 0; it < K_PROP; ++it) {
    zero_kernel<<<(N_NODES * N_CLASS + 255) / 256, blk, 0, stream>>>(agg, N_NODES * N_CLASS);
    scatter_kernel<<<(E + 255) / 256, blk, 0, stream>>>(src, dst, dinv, b, agg, E);
    combine_kernel<<<(N_NODES * N_CLASS + 255) / 256, blk, 0, stream>>>(agg, dinv, x0, b);
  }

  finalize_kernel<<<(N_NODES + 255) / 256, blk, 0, stream>>>(b, (float*)d_out);
}
